// Decoder_64261300682929
// MI455X (gfx1250) — matrix operand order repaired and checked by reading, not compiled, not run on hardware
//
#include <hip/hip_runtime.h>
#include <hip/hip_bf16.h>

// ---------------------------------------------------------------------------
// Types for WMMA (CDNA5 / gfx1250, wave32)
// ---------------------------------------------------------------------------
typedef __attribute__((ext_vector_type(16))) _Float16 v16h;
typedef __attribute__((ext_vector_type(8)))  _Float16 v8h;
typedef __attribute__((ext_vector_type(8)))  float    v8f;

#define B_SZ 32
#define T_ENC 256
#define T_DEC 256
#define ENC 512
#define HID 1024
#define ATT 128
#define CCH 32
#define KCONV 31
#define OUT_D 80
#define PRE 256

// ---------------------------------------------------------------------------
// WMMA fragment loaders.
// A: activations, f16 row-major [M][K].  Contiguous-k A layout used here (16x32):
//   lane l (row = l&15): halves 0..15 = K[16*(l>>4) .. +15]
// B: weights, f16 [N][K] (natural for x @ W.T).  Matching contiguous-k B layout (32x16):
//   lane l (col = l&15): halves 0..15 = K[16*(l>>4) .. +15]  (contiguous in W row)
// ---------------------------------------------------------------------------
__device__ __forceinline__ v16h ld_a_frag(const _Float16* __restrict__ A, int K,
                                          int row, int kk, int lane) {
  const _Float16* p = A + (size_t)row * K + kk + ((lane >> 4) << 4);
  v8h lo = *(const v8h*)(p);
  v8h hi = *(const v8h*)(p + 8);
  v16h a;
#pragma unroll
  for (int i = 0; i < 8; ++i) { a[i] = lo[i]; a[i + 8] = hi[i]; }
  return a;
}

__device__ __forceinline__ v16h ld_b_frag(const _Float16* __restrict__ W, int K,
                                          int col, int kk, int lane) {
  return *(const v16h*)(W + (size_t)col * K + kk + ((lane >> 4) << 4));
}

// ---------------------------------------------------------------------------
// Generic dual-source WMMA GEMM:  C[M][N] = A0@W0.T + A1@W1.T + bias0 + bias1
// One 16x16 output tile per wave, 8 waves (256 threads) per block.
// Grid: (M/16 * N/16) / 8 blocks (always exact for our shapes).
// ---------------------------------------------------------------------------
__global__ void wmma_gemm_dual(const _Float16* __restrict__ A0,
                               const _Float16* __restrict__ W0, int K0,
                               const _Float16* __restrict__ A1,
                               const _Float16* __restrict__ W1, int K1,
                               const float* __restrict__ bias0,
                               const float* __restrict__ bias1,
                               float* __restrict__ C, int M, int N) {
  const int lane = threadIdx.x & 31;
  const int wave = threadIdx.x >> 5;
  const int ntile = N >> 4;
  const int tile = blockIdx.x * 8 + wave;
  const int tm = tile / ntile;
  const int tn = tile - tm * ntile;
  if (tm * 16 >= M) return;  // wave-uniform: EXEC stays all-ones for WMMA

  const int row = tm * 16 + (lane & 15);
  const int col = tn * 16 + (lane & 15);

  v8f acc = {};
  for (int kk = 0; kk < K0; kk += 32) {
    v16h a = ld_a_frag(A0, K0, row, kk, lane);
    v16h b = ld_b_frag(W0, K0, col, kk, lane);
    __builtin_prefetch(W0 + (size_t)col * K0 + kk + 64, 0, 1);
    acc = __builtin_amdgcn_wmma_f32_16x16x32_f16(false, a, false, b,
                                                 (short)0, acc, false, false);
  }
  if (A1) {
    for (int kk = 0; kk < K1; kk += 32) {
      v16h a = ld_a_frag(A1, K1, row, kk, lane);
      v16h b = ld_b_frag(W1, K1, col, kk, lane);
      __builtin_prefetch(W1 + (size_t)col * K1 + kk + 64, 0, 1);
      acc = __builtin_amdgcn_wmma_f32_16x16x32_f16(false, a, false, b,
                                                   (short)0, acc, false, false);
    }
  }
  float bias = 0.f;
  if (bias0) bias += bias0[col];
  if (bias1) bias += bias1[col];

  // C/D layout: VGPR r -> row (tm*16 + 8*(lane>>4) + r), col = tn*16 + lane&15
  const int m0 = tm * 16 + ((lane >> 4) << 3);
#pragma unroll
  for (int r = 0; r < 8; ++r)
    C[(size_t)(m0 + r) * N + col] = acc[r] + bias;
}

// ---------------------------------------------------------------------------
// fp32 -> fp16 conversion (weights / encoder states)
// ---------------------------------------------------------------------------
__global__ void f32_to_f16_kernel(const float* __restrict__ s,
                                  _Float16* __restrict__ d, int n) {
  int i = blockIdx.x * 256 + threadIdx.x;
  if (i < n) d[i] = (_Float16)s[i];
}

__global__ void zero_kernel(unsigned int* __restrict__ p, int nwords) {
  int i = blockIdx.x * 256 + threadIdx.x;
  if (i < nwords) p[i] = 0u;
}

// ---------------------------------------------------------------------------
// Location-sensitive attention: one block per batch row, 256 threads (t index).
// conv(cum) -> tanh energy -> masked softmax -> context; updates cum, writes aw.
// ---------------------------------------------------------------------------
__global__ void attention_kernel(int step,
                                 const float* __restrict__ enc,   // (B,256,512)
                                 const float* __restrict__ pm,    // (B,256,128)
                                 const float* __restrict__ h0,    // (B,1024)
                                 float* __restrict__ cum,         // (B,256)
                                 const int* __restrict__ in_lens,
                                 const float* __restrict__ Ww,    // (128,1024)
                                 const float* __restrict__ Uw,    // (128,32)
                                 const float* __restrict__ Fw,    // (32,1,31)
                                 const float* __restrict__ ww,    // (1,128)
                                 const float* __restrict__ wb,    // (1,)
                                 float* __restrict__ attc,        // (B,512)
                                 float* __restrict__ att_out) {   // (B,256,256)
  const int b = blockIdx.x;
  const int t = threadIdx.x;
  __shared__ float s_prev[256];
  __shared__ float s_aw[256];
  __shared__ float s_hW[128];
  __shared__ float s_red[256];

  const int inl = in_lens[b];
  const float cum_old = (step == 0) ? 0.f : cum[b * 256 + t];
  const float prev =
      (step == 0) ? ((t < inl) ? 1.f / (float)inl : 0.f) : cum_old;
  s_prev[t] = prev;
  __syncthreads();

  // hW = h0 @ W_w.T  (128 dots of length 1024)
  if (t < 128) {
    const float* hr = h0 + b * HID;
    const float* wr = Ww + t * HID;
    float s = 0.f;
    for (int k = 0; k < HID; ++k) s += hr[k] * wr[k];
    s_hW[t] = s;
  }

  // location conv: fc[c] = sum_k prev[t+k-15] * F[c][k]
  float fc[CCH];
#pragma unroll
  for (int c = 0; c < CCH; ++c) {
    const float* fk = Fw + c * KCONV;
    float s = 0.f;
    for (int k = 0; k < KCONV; ++k) {
      int ti = t + k - 15;
      float v = (ti >= 0 && ti < 256) ? s_prev[ti] : 0.f;
      s += v * fk[k];
    }
    fc[c] = s;
  }
  __syncthreads();

  // energy: e = sum_a w[a] * tanh(pm + hW + fc@Uw.T) + wb
  float e = 0.f;
  const float* pmr = pm + (size_t)(b * 256 + t) * ATT;
  for (int a = 0; a < ATT; ++a) {
    const float* ur = Uw + a * CCH;
    float fu = 0.f;
#pragma unroll
    for (int c = 0; c < CCH; ++c) fu += fc[c] * ur[c];
    e += ww[a] * tanhf(pmr[a] + s_hW[a] + fu);
  }
  e += wb[0];
  if (t >= inl) e = -1e9f;

  // masked softmax over t
  s_red[t] = e;
  __syncthreads();
  for (int s2 = 128; s2 > 0; s2 >>= 1) {
    if (t < s2) s_red[t] = fmaxf(s_red[t], s_red[t + s2]);
    __syncthreads();
  }
  const float mx = s_red[0];
  __syncthreads();
  const float ex = __expf(e - mx);
  s_red[t] = ex;
  __syncthreads();
  for (int s2 = 128; s2 > 0; s2 >>= 1) {
    if (t < s2) s_red[t] += s_red[t + s2];
    __syncthreads();
  }
  const float aw = ex / s_red[0];
  s_aw[t] = aw;
  cum[b * 256 + t] = cum_old + aw;
  att_out[(size_t)(b * 256 + step) * 256 + t] = aw;
  __syncthreads();

  // context: att_c[e] = sum_t aw[t] * enc[b][t][e]
  for (int e2 = t; e2 < ENC; e2 += 256) {
    float s = 0.f;
    for (int tt = 0; tt < 256; ++tt)
      s += s_aw[tt] * enc[(size_t)(b * 256 + tt) * ENC + e2];
    attc[b * ENC + e2] = s;
  }
}

// ---------------------------------------------------------------------------
// Prenet (two relu+dropout-mask layers) + build xin f16 = [att_c | p].
// Single block, 256 threads (one output channel each, loop over batch).
// ---------------------------------------------------------------------------
__global__ void prenet_kernel(int step,
                              const float* __restrict__ dect,   // (B,256,80)
                              const float* __restrict__ masks,  // (256,2,B,256)
                              const float* __restrict__ w0, const float* __restrict__ b0,
                              const float* __restrict__ w1, const float* __restrict__ b1,
                              const float* __restrict__ attc,   // (B,512)
                              _Float16* __restrict__ xin) {     // (B,768)
  const int j = threadIdx.x;
  __shared__ float s_p0[B_SZ][PRE];

  const float* wr0 = w0 + j * OUT_D;
  for (int b = 0; b < B_SZ; ++b) {
    float s = b0[j];
    if (step > 0) {
      const float* x = dect + (size_t)(b * T_DEC + step - 1) * OUT_D;
      for (int k = 0; k < OUT_D; ++k) s += x[k] * wr0[k];
    }
    s = fmaxf(s, 0.f) * masks[((size_t)(step * 2 + 0) * B_SZ + b) * PRE + j];
    s_p0[b][j] = s;
  }
  __syncthreads();

  const float* wr1 = w1 + j * PRE;
  for (int b = 0; b < B_SZ; ++b) {
    float s = b1[j];
    for (int k = 0; k < PRE; ++k) s += s_p0[b][k] * wr1[k];
    s = fmaxf(s, 0.f) * masks[((size_t)(step * 2 + 1) * B_SZ + b) * PRE + j];
    xin[b * (ENC + PRE) + ENC + j] = (_Float16)s;
  }
  // att_c portion of xin
  for (int i = j; i < B_SZ * ENC; i += 256) {
    int b = i >> 9, e = i & 511;
    xin[b * (ENC + PRE) + e] = (_Float16)attc[i];
  }
}

// ---------------------------------------------------------------------------
// LSTM cell + zoneout.  idx over B*HID.
// gates layout (B,4096): [i | f | g | o] chunks of 1024.
// ---------------------------------------------------------------------------
__device__ __forceinline__ float sigf(float x) { return 1.f / (1.f + __expf(-x)); }

__global__ void lstm_cell_kernel(const float* __restrict__ gates,
                                 float* __restrict__ h, float* __restrict__ c,
                                 _Float16* __restrict__ h16) {
  const int idx = blockIdx.x * 256 + threadIdx.x;  // 0..32767
  const int b = idx >> 10, j = idx & 1023;
  const float* g = gates + (size_t)b * 4 * HID;
  const float gi = g[j], gf = g[j + HID], gg = g[j + 2 * HID], go = g[j + 3 * HID];
  const float co = c[idx], ho = h[idx];
  const float cn = sigf(gf) * co + sigf(gi) * tanhf(gg);
  const float hn = sigf(go) * tanhf(cn);
  const float h2 = 0.1f * ho + 0.9f * hn;  // zoneout 0.1
  const float c2 = 0.1f * co + 0.9f * cn;
  h[idx] = h2;
  c[idx] = c2;
  h16[idx] = (_Float16)h2;
}

// ---------------------------------------------------------------------------
// Output projection: out = [h1|att_c] @ feat_w.T ; logit = [h1|att_c] @ prob_w.T
// One block per batch row; threads 0..79 -> feature outputs, thread 80 -> logit.
// ---------------------------------------------------------------------------
__global__ void output_kernel(int step, const float* __restrict__ h1,
                              const float* __restrict__ attc,
                              const float* __restrict__ featw,  // (80,1536)
                              const float* __restrict__ probw,  // (1,1536)
                              const float* __restrict__ probb,
                              float* __restrict__ out) {
  const int b = blockIdx.x, o = threadIdx.x;
  const float* hr = h1 + b * HID;
  const float* ar = attc + b * ENC;
  if (o < OUT_D) {
    const float* w = featw + o * (ENC + HID);
    float s = 0.f;
    for (int k = 0; k < HID; ++k) s += hr[k] * w[k];
    for (int k = 0; k < ENC; ++k) s += ar[k] * w[HID + k];
    out[(size_t)(b * OUT_D + o) * T_DEC + step] = s;
  } else if (o == OUT_D) {
    float s = probb[0];
    for (int k = 0; k < HID; ++k) s += hr[k] * probw[k];
    for (int k = 0; k < ENC; ++k) s += ar[k] * probw[HID + k];
    out[(size_t)B_SZ * OUT_D * T_DEC + b * T_DEC + step] = s;
  }
}

// ---------------------------------------------------------------------------
// Host orchestration
// ---------------------------------------------------------------------------
extern "C" void kernel_launch(void* const* d_in, const int* in_sizes, int n_in,
                              void* d_out, int out_size, void* d_ws, size_t ws_size,
                              hipStream_t stream) {
  (void)in_sizes; (void)n_in; (void)out_size; (void)ws_size;
  const float* enc    = (const float*)d_in[0];
  const float* dect   = (const float*)d_in[1];
  const float* masks  = (const float*)d_in[2];
  const int*   inlens = (const int*)d_in[3];
  const float* Vw     = (const float*)d_in[4];
  const float* Vb     = (const float*)d_in[5];
  const float* Ww     = (const float*)d_in[6];
  const float* Uw     = (const float*)d_in[7];
  const float* Fw     = (const float*)d_in[8];
  const float* ww     = (const float*)d_in[9];
  const float* wb     = (const float*)d_in[10];
  const float* pw0    = (const float*)d_in[11];
  const float* pb0    = (const float*)d_in[12];
  const float* pw1    = (const float*)d_in[13];
  const float* pb1    = (const float*)d_in[14];
  const float* l0wih  = (const float*)d_in[15];
  const float* l0whh  = (const float*)d_in[16];
  const float* l0bih  = (const float*)d_in[17];
  const float* l0bhh  = (const float*)d_in[18];
  const float* l1wih  = (const float*)d_in[19];
  const float* l1whh  = (const float*)d_in[20];
  const float* l1bih  = (const float*)d_in[21];
  const float* l1bhh  = (const float*)d_in[22];
  const float* featw  = (const float*)d_in[23];
  const float* probw  = (const float*)d_in[24];
  const float* probb  = (const float*)d_in[25];
  float* out = (float*)d_out;

  // ---- workspace carving (256B aligned) ----
  char* cur = (char*)d_ws;
  auto alloc = [&](size_t bytes) -> char* {
    char* p = cur;
    cur += (bytes + 255) & ~(size_t)255;
    return p;
  };
  const int N_ENC16 = B_SZ * T_ENC * ENC;       // 4,194,304
  const int N_W0IH  = 4 * HID * (ENC + PRE);    // 3,145,728
  const int N_WHH   = 4 * HID * HID;            // 4,194,304
  const int N_VW    = ATT * ENC;                // 65,536
  _Float16* enc16  = (_Float16*)alloc((size_t)N_ENC16 * 2);
  _Float16* w0ih16 = (_Float16*)alloc((size_t)N_W0IH * 2);
  _Float16* w0hh16 = (_Float16*)alloc((size_t)N_WHH * 2);
  _Float16* w1ih16 = (_Float16*)alloc((size_t)N_WHH * 2);
  _Float16* w1hh16 = (_Float16*)alloc((size_t)N_WHH * 2);
  _Float16* vw16   = (_Float16*)alloc((size_t)N_VW * 2);
  float*    pm     = (float*)alloc((size_t)B_SZ * T_ENC * ATT * 4);
  // contiguous per-call state block (zeroed once per call, deterministic)
  const size_t state_bytes = 4 * (size_t)B_SZ * HID * 4     // h0,c0,h1,c1 f32
                           + 2 * (size_t)B_SZ * HID * 2     // h0_f16,h1_f16
                           + (size_t)B_SZ * T_ENC * 4;      // cum
  char* st = alloc(state_bytes);
  float*    h0   = (float*)st;
  float*    c0   = h0 + B_SZ * HID;
  float*    h1   = c0 + B_SZ * HID;
  float*    c1   = h1 + B_SZ * HID;
  _Float16* h016 = (_Float16*)(c1 + B_SZ * HID);
  _Float16* h116 = h016 + B_SZ * HID;
  float*    cum  = (float*)(h116 + B_SZ * HID);
  float*    attc  = (float*)alloc((size_t)B_SZ * ENC * 4);
  _Float16* xin16 = (_Float16*)alloc((size_t)B_SZ * (ENC + PRE) * 2);
  float*    gates = (float*)alloc((size_t)B_SZ * 4 * HID * 4);

  // ---- one-time setup ----
  auto cvt = [&](const float* s, _Float16* d, int n) {
    f32_to_f16_kernel<<<(n + 255) / 256, 256, 0, stream>>>(s, d, n);
  };
  cvt(enc,   enc16,  N_ENC16);
  cvt(l0wih, w0ih16, N_W0IH);
  cvt(l0whh, w0hh16, N_WHH);
  cvt(l1wih, w1ih16, N_WHH);
  cvt(l1whh, w1hh16, N_WHH);
  cvt(Vw,    vw16,   N_VW);

  const int zero_words = (int)(state_bytes / 4);
  zero_kernel<<<(zero_words + 255) / 256, 256, 0, stream>>>((unsigned int*)st, zero_words);

  // processed_memory = enc @ V_w.T + V_b   (M=8192, K=512, N=128)
  {
    const int M = B_SZ * T_ENC, N = ATT;
    const int tiles = (M / 16) * (N / 16);  // 4096
    wmma_gemm_dual<<<tiles / 8, 256, 0, stream>>>(
        enc16, vw16, ENC, nullptr, nullptr, 0, Vb, nullptr, pm, M, N);
  }

  float* att_out = out + (size_t)B_SZ * OUT_D * T_DEC + (size_t)B_SZ * T_DEC;

  // ---- sequential decode ----
  for (int step = 0; step < T_DEC; ++step) {
    attention_kernel<<<B_SZ, 256, 0, stream>>>(
        step, enc, pm, h0, cum, inlens, Ww, Uw, Fw, ww, wb, attc, att_out);

    prenet_kernel<<<1, 256, 0, stream>>>(
        step, dect, masks, pw0, pb0, pw1, pb1, attc, xin16);

    // LSTM0 gates: xin(768) @ w0ih.T + h0(1024) @ w0hh.T   (M=32, N=4096)
    {
      const int tiles = (B_SZ / 16) * (4 * HID / 16);  // 512
      wmma_gemm_dual<<<tiles / 8, 256, 0, stream>>>(
          xin16, w0ih16, ENC + PRE, h016, w0hh16, HID,
          l0bih, l0bhh, gates, B_SZ, 4 * HID);
    }
    lstm_cell_kernel<<<(B_SZ * HID) / 256, 256, 0, stream>>>(gates, h0, c0, h016);

    // LSTM1 gates: h0(1024) @ w1ih.T + h1(1024) @ w1hh.T
    {
      const int tiles = (B_SZ / 16) * (4 * HID / 16);
      wmma_gemm_dual<<<tiles / 8, 256, 0, stream>>>(
          h016, w1ih16, HID, h116, w1hh16, HID,
          l1bih, l1bhh, gates, B_SZ, 4 * HID);
    }
    lstm_cell_kernel<<<(B_SZ * HID) / 256, 256, 0, stream>>>(gates, h1, c1, h116);

    output_kernel<<<B_SZ, 128, 0, stream>>>(step, h1, attc, featw, probw, probb, out);
  }
}
